// RBFKernel_65481071396583
// MI455X (gfx1250) — matrix operand order repaired and checked by reading, not compiled, not run on hardware
//
#include <hip/hip_runtime.h>

// ---------- types ----------
typedef __bf16 v16bf __attribute__((ext_vector_type(16)));
typedef __bf16 v8bf  __attribute__((ext_vector_type(8)));
typedef __bf16 v4bf  __attribute__((ext_vector_type(4)));
typedef float  v8f   __attribute__((ext_vector_type(8)));

#define N_ROWS 8192   // rows of X (M) and rows of Y (N)
#define D_DIM  256    // feature dim (K)
#define BM     128
#define BN     128
#define LDW    264    // 256 + 8 bf16 padding -> 528B row stride, conflict-free
#define SMEM_BYTES (4 * BM * LDW * 2)   // Xhi,Xlo,Yhi,Ylo = 270336 B (< 320KB WGP LDS)

// ---------- helpers ----------
__device__ __forceinline__ __bf16 f32_to_bf16_rne(float f) {
    unsigned u = __builtin_bit_cast(unsigned, f);
    u += 0x7FFFu + ((u >> 16) & 1u);            // round-to-nearest-even
    unsigned short h = (unsigned short)(u >> 16);
    return __builtin_bit_cast(__bf16, h);
}

__device__ __forceinline__ void split_bf16(float f, __bf16& hi, __bf16& lo) {
    hi = f32_to_bf16_rne(f);
    lo = f32_to_bf16_rne(f - (float)hi);        // residual, exact subtraction
}

// A fragment (16x32 bf16, MxK): lane<16 -> K = [ka..ka+7] and [ka+16..ka+23]
__device__ __forceinline__ v16bf load_a_frag(const __bf16* rowp, int ka) {
    v8bf a0 = *(const v8bf*)(rowp + ka);
    v8bf a1 = *(const v8bf*)(rowp + ka + 16);
    return __builtin_shufflevector(a0, a1, 0,1,2,3,4,5,6,7,8,9,10,11,12,13,14,15);
}

// B fragment (32x16 bf16, KxN): lane<16 -> K = [kb..kb+7] and [kb+16..kb+23]
__device__ __forceinline__ v16bf load_b_frag(const __bf16* rowp, int kb) {
    v8bf b0 = *(const v8bf*)(rowp + kb);
    v8bf b1 = *(const v8bf*)(rowp + kb + 16);
    return __builtin_shufflevector(b0, b1, 0,1,2,3,4,5,6,7,8,9,10,11,12,13,14,15);
}

// ---------- pass 1: row norms ----------
__global__ void __launch_bounds__(256)
rbf_row_norms(const float* __restrict__ x, const float* __restrict__ y,
              float* __restrict__ xs, float* __restrict__ ys) {
    int r = blockIdx.x * blockDim.x + threadIdx.x;       // 0..8191
    const float4* xr = (const float4*)(x + (size_t)r * D_DIM);
    const float4* yr = (const float4*)(y + (size_t)r * D_DIM);
    float sx = 0.f, sy = 0.f;
    #pragma unroll 8
    for (int i = 0; i < D_DIM / 4; ++i) {
        float4 a = xr[i], b = yr[i];
        sx += a.x*a.x + a.y*a.y + a.z*a.z + a.w*a.w;
        sy += b.x*b.x + b.y*b.y + b.z*b.z + b.w*b.w;
    }
    xs[r] = sx;
    ys[r] = sy;
}

// ---------- pass 2: WMMA GEMM + RBF epilogue ----------
__global__ void __launch_bounds__(256, 1)
rbf_wmma_kernel(const float* __restrict__ X, const float* __restrict__ Y,
                const float* __restrict__ xs, const float* __restrict__ ys,
                float* __restrict__ out) {
    extern __shared__ __bf16 smem[];
    __bf16* Xhi = smem;
    __bf16* Xlo = Xhi + BM * LDW;
    __bf16* Yhi = Xlo + BM * LDW;
    __bf16* Ylo = Yhi + BN * LDW;

    const int tid = threadIdx.x;
    const int bx = blockIdx.x;    // N tile
    const int by = blockIdx.y;    // M tile

    // ---- stage both fp32 tiles, split into bf16 hi/lo in LDS ----
    const float* gx = X + (size_t)by * BM * D_DIM;
    const float* gy = Y + (size_t)bx * BN * D_DIM;
    #pragma unroll
    for (int it = 0; it < (BM * D_DIM / 4) / 256; ++it) {   // 32 iters
        int idx4 = it * 256 + tid;          // float4 index within tile
        int row  = idx4 >> 6;               // 64 float4 per row
        int col  = (idx4 & 63) << 2;
        float4 vx = *(const float4*)(gx + row * D_DIM + col);
        float4 vy = *(const float4*)(gy + row * D_DIM + col);

        __bf16 h0,h1,h2,h3,l0,l1,l2,l3;
        split_bf16(vx.x, h0, l0); split_bf16(vx.y, h1, l1);
        split_bf16(vx.z, h2, l2); split_bf16(vx.w, h3, l3);
        *(v4bf*)(Xhi + row * LDW + col) = (v4bf){h0,h1,h2,h3};
        *(v4bf*)(Xlo + row * LDW + col) = (v4bf){l0,l1,l2,l3};

        split_bf16(vy.x, h0, l0); split_bf16(vy.y, h1, l1);
        split_bf16(vy.z, h2, l2); split_bf16(vy.w, h3, l3);
        *(v4bf*)(Yhi + row * LDW + col) = (v4bf){h0,h1,h2,h3};
        *(v4bf*)(Ylo + row * LDW + col) = (v4bf){l0,l1,l2,l3};
    }
    __syncthreads();

    // ---- wave tiling: 8 waves, 2 (M) x 4 (N); wave patch = 64x32 ----
    const int lane  = tid & 31;
    const int wid   = tid >> 5;
    const int half  = lane >> 4;      // 0: lanes 0-15, 1: lanes 16-31
    const int l16   = lane & 15;
    const int waveM = wid & 1;        // 0..1 -> row offset 64*waveM
    const int waveN = wid >> 1;       // 0..3 -> col offset 32*waveN

    v8f acc[4][2];
    #pragma unroll
    for (int mt = 0; mt < 4; ++mt)
        #pragma unroll
        for (int nt = 0; nt < 2; ++nt)
            acc[mt][nt] = (v8f)(0.0f);

    #pragma unroll
    for (int kc = 0; kc < D_DIM / 32; ++kc) {       // 8 K-chunks
        const int k0 = kc * 32;
        const int ka = k0 + half * 8;               // A-layout K offset
        const int kb = k0 + half * 8;               // B-layout K offset

        v16bf ah[4], al[4];
        #pragma unroll
        for (int mt = 0; mt < 4; ++mt) {
            int r = (waveM * 64 + mt * 16 + l16) * LDW;
            ah[mt] = load_a_frag(Xhi + r, ka);
            al[mt] = load_a_frag(Xlo + r, ka);
        }
        v16bf bh[2], bl[2];
        #pragma unroll
        for (int nt = 0; nt < 2; ++nt) {
            int r = (waveN * 32 + nt * 16 + l16) * LDW;
            bh[nt] = load_b_frag(Yhi + r, kb);
            bl[nt] = load_b_frag(Ylo + r, kb);
        }

        #pragma unroll
        for (int mt = 0; mt < 4; ++mt) {
            #pragma unroll
            for (int nt = 0; nt < 2; ++nt) {
                // bf16x3: hi*hi + hi*lo + lo*hi, fp32 accumulate
                acc[mt][nt] = __builtin_amdgcn_wmma_f32_16x16x32_bf16(
                    false, ah[mt], false, bh[nt], (short)0, acc[mt][nt], false, false);
                acc[mt][nt] = __builtin_amdgcn_wmma_f32_16x16x32_bf16(
                    false, ah[mt], false, bl[nt], (short)0, acc[mt][nt], false, false);
                acc[mt][nt] = __builtin_amdgcn_wmma_f32_16x16x32_bf16(
                    false, al[mt], false, bh[nt], (short)0, acc[mt][nt], false, false);
            }
        }
    }

    // ---- epilogue: dist^2 = xs + ys - 2*cross ; out = exp(-dist^2) ----
    #pragma unroll
    for (int mt = 0; mt < 4; ++mt) {
        // C/D layout: VGPR v, lane<16 -> M = v ; lane>=16 -> M = v+8 ; N = lane&15
        int gm0 = by * BM + waveM * 64 + mt * 16 + half * 8;
        float xsv[8];
        #pragma unroll
        for (int v = 0; v < 8; ++v) xsv[v] = xs[gm0 + v];

        #pragma unroll
        for (int nt = 0; nt < 2; ++nt) {
            int gn = bx * BN + waveN * 32 + nt * 16 + l16;
            float ysq = ys[gn];
            v8f a = acc[mt][nt];
            #pragma unroll
            for (int v = 0; v < 8; ++v) {
                float d = xsv[v] + ysq - 2.0f * a[v];
                d = fmaxf(d, 0.0f);
                out[(size_t)(gm0 + v) * N_ROWS + gn] = __expf(-d);  // GAMMA = 1.0
            }
        }
    }
}

// ---------- launcher ----------
extern "C" void kernel_launch(void* const* d_in, const int* in_sizes, int n_in,
                              void* d_out, int out_size, void* d_ws, size_t ws_size,
                              hipStream_t stream) {
    const float* x = (const float*)d_in[0];
    const float* y = (const float*)d_in[1];
    float* out = (float*)d_out;

    float* xs = (float*)d_ws;           // 8192 floats
    float* ys = xs + N_ROWS;            // 8192 floats (64 KB total in d_ws)

    rbf_row_norms<<<N_ROWS / 256, 256, 0, stream>>>(x, y, xs, ys);

    dim3 grid(N_ROWS / BN, N_ROWS / BM);   // 64 x 64 blocks
    rbf_wmma_kernel<<<grid, 256, SMEM_BYTES, stream>>>(x, y, xs, ys, out);
}
